// QGCL_14516989461122
// MI455X (gfx1250) — matrix operand order repaired and checked by reading, not compiled, not run on hardware
//
#include <hip/hip_runtime.h>
#include <hip/hip_bf16.h>
#include <hip/hip_fp16.h>

typedef __attribute__((ext_vector_type(16))) _Float16 v16h;
typedef __attribute__((ext_vector_type(8)))  float    v8f;

#define NN 50000
#define NE 800000
#define DF 64
#define HID 64
#define NORM_INV 0.01f
#define FRAG_H 512  // halves per packed fragment: 32 lanes x 16 f16

__device__ __forceinline__ float silu_f(float x) { return x / (1.f + __expf(-x)); }

// 16-bit B-matrix K-mapping (matches A fill): elems 0-7 K=8*hi+s, elems 8-15 K=16+8*hi+(s-8)
__device__ __forceinline__ int kpairB(int v, int hi) { return 2 * v + 8 * hi + ((v >= 4) ? 8 : 0); }

__device__ __forceinline__ v8f wmma_fp16(v16h a, v16h b, v8f c) {
    return __builtin_amdgcn_wmma_f32_16x16x32_f16(false, a, false, b, (short)0, c, false, false);
}

// packed B fragment: 32 B per lane (2x b128, global or LDS depending on pointer)
__device__ __forceinline__ v16h b_frag_pk(const _Float16* P, int fi, int lane) {
    return *(const v16h*)(P + ((size_t)fi * 32 + lane) * 16);
}

// pack two contiguous 8-float spans into the 16-bit A-fragment element order
__device__ __forceinline__ v16h pack16(float4 p0, float4 p1, float4 p2, float4 p3) {
    v16h a;
    a[0]  = (_Float16)p0.x; a[1]  = (_Float16)p0.y; a[2]  = (_Float16)p0.z; a[3]  = (_Float16)p0.w;
    a[4]  = (_Float16)p1.x; a[5]  = (_Float16)p1.y; a[6]  = (_Float16)p1.z; a[7]  = (_Float16)p1.w;
    a[8]  = (_Float16)p2.x; a[9]  = (_Float16)p2.y; a[10] = (_Float16)p2.z; a[11] = (_Float16)p2.w;
    a[12] = (_Float16)p3.x; a[13] = (_Float16)p3.y; a[14] = (_Float16)p3.z; a[15] = (_Float16)p3.w;
    return a;
}

// A fragment from an LDS tile: lane's halves = spans [kbase+8*hi,+8) and [kbase+16+8*hi,+8)
template <int LDA>
__device__ __forceinline__ v16h a_frag_lds(const float (*A)[LDA], int kbase, int halfLane, int hi) {
    const float* rp = A[halfLane] + kbase + 8 * hi;
    float4 p0 = *(const float4*)(rp + 0);
    float4 p1 = *(const float4*)(rp + 4);
    float4 p2 = *(const float4*)(rp + 16);
    float4 p3 = *(const float4*)(rp + 20);
    return pack16(p0, p1, p2, p3);
}

// --- CDNA5 async global->LDS (ASYNCcnt-tracked); LDS offset = low 32 bits of generic ptr ---
__device__ __forceinline__ void async_load_b128(unsigned lds_byte_addr, const void* gaddr) {
    asm volatile("global_load_async_to_lds_b128 %0, %1, off"
                 :: "v"(lds_byte_addr), "v"((unsigned long long)(uintptr_t)gaddr)
                 : "memory");
}
__device__ __forceinline__ void wait_async0() {
    asm volatile("s_wait_asynccnt 0x0" ::: "memory");
}

// ---------------- quantum circuit: 3-qubit statevector, fully register-resident ----------------
__device__ __forceinline__ void rx_layer3(float* sr, float* si, float cb, float sb) {
    #pragma unroll
    for (int q = 0; q < 3; ++q) {
        int stride = 4 >> q;
        #pragma unroll
        for (int i = 0; i < 8; ++i) {
            if ((i & stride) == 0) {
                int j = i | stride;
                float a0r = sr[i], a0i = si[i], a1r = sr[j], a1i = si[j];
                sr[i] = cb * a0r + sb * a1i;  si[i] = cb * a0i - sb * a1r;
                sr[j] = sb * a0i + cb * a1r;  si[j] = cb * a1i - sb * a0r;
            }
        }
    }
}

__device__ void qcirc(const float* x, float alpha, float beta, float gamma, float delta,
                      const float* __restrict__ Lam, float* ez) {
    float sr[8], si[8];
    #pragma unroll
    for (int i = 0; i < 8; ++i) { sr[i] = 0.f; si[i] = 0.f; }
    sr[0] = 1.f;

    #pragma unroll
    for (int q = 0; q < 3; ++q) {  // RY(x*alpha)
        float th = 0.5f * x[q] * alpha;
        float cc = __cosf(th), ss = __sinf(th);
        int stride = 4 >> q;
        #pragma unroll
        for (int i = 0; i < 8; ++i) {
            if ((i & stride) == 0) {
                int j = i | stride;
                float a0r = sr[i], a0i = si[i], a1r = sr[j], a1i = si[j];
                sr[i] = cc * a0r - ss * a1r;  si[i] = cc * a0i - ss * a1i;
                sr[j] = ss * a0r + cc * a1r;  si[j] = ss * a0i + cc * a1i;
            }
        }
    }
    #pragma unroll
    for (int qi = 0; qi < 3; ++qi) {  // IsingZZ diagonal phases
        #pragma unroll
        for (int qj = qi + 1; qj < 3; ++qj) {
            float half = 0.25f * gamma * (Lam[qi * 3 + qj] + Lam[qj * 3 + qi]);
            float ch = __cosf(half), sh = __sinf(half);
            int bi = 4 >> qi, bj = 4 >> qj;
            #pragma unroll
            for (int i = 0; i < 8; ++i) {
                float zz = ((i & bi) ? -1.f : 1.f) * ((i & bj) ? -1.f : 1.f);
                float pim = -zz * sh;
                float nr = sr[i] * ch - si[i] * pim;
                float ni = sr[i] * pim + si[i] * ch;
                sr[i] = nr; si[i] = ni;
            }
        }
    }
    float cb = __cosf(0.5f * beta), sb = __sinf(0.5f * beta);
    rx_layer3(sr, si, cb, sb);
    const float IS2 = 0.70710678118654752f;
    #pragma unroll
    for (int q = 0; q < 3; ++q) {  // anharmonic: RZ,H,RZ,H (MU=0.5)
        float xq2 = x[q] * x[q];
        int stride = 4 >> q;
        #pragma unroll
        for (int pass = 0; pass < 2; ++pass) {
            float t = pass ? (delta * xq2) : (delta * (1.f - 0.5f * xq2));
            float cz = __cosf(0.5f * t), sz = __sinf(0.5f * t);
            #pragma unroll
            for (int i = 0; i < 8; ++i) {
                float pim = (i & stride) ? sz : -sz;
                float nr = sr[i] * cz - si[i] * pim;
                float ni = sr[i] * pim + si[i] * cz;
                sr[i] = nr; si[i] = ni;
            }
            #pragma unroll
            for (int i = 0; i < 8; ++i) {
                if ((i & stride) == 0) {
                    int j = i | stride;
                    float a0r = sr[i], a0i = si[i], a1r = sr[j], a1i = si[j];
                    sr[i] = (a0r + a1r) * IS2;  si[i] = (a0i + a1i) * IS2;
                    sr[j] = (a0r - a1r) * IS2;  si[j] = (a0i - a1i) * IS2;
                }
            }
        }
    }
    rx_layer3(sr, si, cb, sb);
    #pragma unroll
    for (int q = 0; q < 3; ++q) {
        int stride = 4 >> q;
        float e = 0.f;
        #pragma unroll
        for (int i = 0; i < 8; ++i) {
            float p = sr[i] * sr[i] + si[i] * si[i];
            e += (i & stride) ? -p : p;
        }
        ez[q] = e;
    }
}

// ---------------- zero the aggregation workspace (graph replays re-accumulate) ----------------
__global__ void qgcl_zero_kernel(float4* __restrict__ p, int n4) {
    int i = blockIdx.x * blockDim.x + threadIdx.x;
    if (i < n4) p[i] = make_float4(0.f, 0.f, 0.f, 0.f);
}

// ---------------- weight pre-pack: f32 row-major -> fragment-ordered f16 (lane-major) ----------
__global__ __launch_bounds__(32)
void qgcl_pack_kernel(const float* __restrict__ W, int krows, int ncols, int NT,
                      float rowScale, int scaleRow0, int scaleRow1,
                      _Float16* __restrict__ dst) {
    int fi = blockIdx.x, lane = threadIdx.x;
    int kc = fi / NT, nt = fi % NT;
    int hi = lane >> 4, halfLane = lane & 15;
    int n = nt * 16 + halfLane;
    _Float16* o = dst + ((size_t)fi * 32 + lane) * 16;
    #pragma unroll
    for (int v = 0; v < 8; ++v) {
        #pragma unroll
        for (int p = 0; p < 2; ++p) {
            int k = kc * 32 + kpairB(v, hi) + p;
            float w = (n < ncols && k < krows) ? W[(size_t)k * ncols + n] : 0.f;
            if (k >= scaleRow0 && k < scaleRow1) w *= rowScale;
            o[2 * v + p] = (_Float16)w;
        }
    }
}

// ---------------- edge model: mij = silu(silu([h_r|h_c]@eW1+eb1)@eW2+eb2); scatter-add --------
__global__ __launch_bounds__(256)
void qgcl_edge_kernel(const float* __restrict__ h, const int* __restrict__ ei,
                      const _Float16* __restrict__ packE1, const float* __restrict__ eb1,
                      const float* __restrict__ eb2,
                      float* __restrict__ mij, float* __restrict__ agg) {
    __shared__ __align__(16) float act[8][16][64];     // 32 KB per-wave activation tiles
    __shared__ __align__(32) _Float16 wlds[24 * FRAG_H];  // 24 KB packed eW1(16)+eW2(8) frags
    __shared__ int rows_s[8][16];

    const int wave = threadIdx.x >> 5;
    const int lane = threadIdx.x & 31;
    const int halfLane = lane & 15;
    const int hi = lane >> 4;

    // async-stage the 24KB packed edge-weight block into LDS (48 b128 async instrs / block)
    {
        unsigned ldsW = (unsigned)(uintptr_t)&wlds[0];
        #pragma unroll
        for (int r = 0; r < 6; ++r) {
            unsigned off = (unsigned)(r * 256 + threadIdx.x) * 16u;
            async_load_b128(ldsW + off, (const char*)packE1 + off);
        }
    }

    const int e0 = (blockIdx.x * 8 + wave) * 16;
    const int em = e0 + halfLane;
    const int r = ei[em];
    const int c = ei[NE + em];
    rows_s[wave][halfLane] = r;
    const float* hr = h + (size_t)r * DF;
    const float* hc = h + (size_t)c * DF;

    const _Float16* WL1 = wlds;                 // 16 fragments
    const _Float16* WL2 = wlds + 16 * FRAG_H;   // 8 fragments

    wait_async0();
    __syncthreads();

    const v8f z8 = {0.f, 0.f, 0.f, 0.f, 0.f, 0.f, 0.f, 0.f};
    v8f acc[4] = {z8, z8, z8, z8};

    // layer 1: [16,128] @ [128,64]; A rows gathered as two 32B spans per chunk (4x b128)
    #pragma unroll
    for (int kc = 0; kc < 4; ++kc) {
        const float* base = ((kc < 2) ? hr : hc) + (kc & 1) * 32 + 8 * hi;
        float4 p0 = *(const float4*)(base + 0);
        float4 p1 = *(const float4*)(base + 4);
        float4 p2 = *(const float4*)(base + 16);
        float4 p3 = *(const float4*)(base + 20);
        v16h a = pack16(p0, p1, p2, p3);
        #pragma unroll
        for (int nt = 0; nt < 4; ++nt)
            acc[nt] = wmma_fp16(a, b_frag_pk(WL1, kc * 4 + nt, lane), acc[nt]);
    }

    float (*A)[64] = act[wave];
    #pragma unroll
    for (int nt = 0; nt < 4; ++nt) {
        int n = nt * 16 + halfLane;
        float bias = eb1[n];
        #pragma unroll
        for (int rr = 0; rr < 8; ++rr)
            A[rr + 8 * hi][n] = silu_f(acc[nt][rr] + bias);  // D layout: M=r+8*hi, N=halfLane
    }
    __syncthreads();

    // layer 2: [16,64] @ [64,64]
    v16h a2c0 = a_frag_lds<64>(A, 0, halfLane, hi);
    v16h a2c1 = a_frag_lds<64>(A, 32, halfLane, hi);
    v8f acc2[4] = {z8, z8, z8, z8};
    #pragma unroll
    for (int nt = 0; nt < 4; ++nt) {
        acc2[nt] = wmma_fp16(a2c0, b_frag_pk(WL2, nt, lane), acc2[nt]);
        acc2[nt] = wmma_fp16(a2c1, b_frag_pk(WL2, 4 + nt, lane), acc2[nt]);
    }
    __syncthreads();
    #pragma unroll
    for (int nt = 0; nt < 4; ++nt) {
        int n = nt * 16 + halfLane;
        float bias = eb2[n];
        #pragma unroll
        for (int rr = 0; rr < 8; ++rr)
            A[rr + 8 * hi][n] = silu_f(acc2[nt][rr] + bias);
    }
    __syncthreads();

    // coalesced mij writeback: 16 edge rows x 64 f32 = 256 float4
    float4* mo = (float4*)(mij + (size_t)e0 * HID);
    const float4* Af4 = (const float4*)&A[0][0];
    #pragma unroll
    for (int i = 0; i < 8; ++i)
        mo[lane + 32 * i] = Af4[lane + 32 * i];

    // scatter-add (un-normalized; 1/NORM folded into packed qW1/pW1); 2 lanes per edge
    int m = lane >> 1;
    int foff = (lane & 1) * 32;
    float* ap = agg + (size_t)rows_s[wave][m] * HID + foff;
    const float* As = &A[m][foff];
    #pragma unroll
    for (int j = 0; j < 32; ++j)
        atomicAdd(ap + j, As[j]);
}

// ---------------- node model: q-net -> quantum circuit -> p-net + residual --------------------
__global__ __launch_bounds__(128)
void qgcl_node_kernel(const float* __restrict__ h, const float* __restrict__ agg,
                      const _Float16* __restrict__ packQ1, const float* __restrict__ qb1,
                      const _Float16* __restrict__ packQ2, const float* __restrict__ qb2,
                      const _Float16* __restrict__ packP1, const float* __restrict__ pb1,
                      const _Float16* __restrict__ packP2, const float* __restrict__ pb2,
                      const float* __restrict__ alpha_p, const float* __restrict__ beta_p,
                      const float* __restrict__ gamma_p, const float* __restrict__ delta_p,
                      const float* __restrict__ Lam, float* __restrict__ out) {
    __shared__ __align__(16) float a2s[4][16][160];  // [h | agg(raw) | q_out | zero-pad to 160]
    __shared__ __align__(16) float hids[4][16][64];
    __shared__ float qins[4][16][4];

    const int wave = threadIdx.x >> 5;
    const int lane = threadIdx.x & 31;
    const int halfLane = lane & 15;
    const int hi = lane >> 4;
    const int nb = (blockIdx.x * 4 + wave) * 16;

    float (*A2)[160] = a2s[wave];
    float (*HD)[64] = hids[wave];
    float (*QI)[4] = qins[wave];

    // async-stage a = [h | agg] into LDS (raw agg: 1/NORM folded into packed weights)
    unsigned ldsA2 = (unsigned)(uintptr_t)&A2[0][0];
    #pragma unroll
    for (int i = 0; i < 8; ++i) {
        int idx4 = lane + 32 * i;           // 0..255 over 16 rows x 16 float4
        int m = idx4 >> 4, f4 = idx4 & 15;
        int node = nb + m; if (node >= NN) node = NN - 1;
        unsigned dst = ldsA2 + (unsigned)(m * 160 + f4 * 4) * 4u;
        async_load_b128(dst,            h   + (size_t)node * DF  + f4 * 4);
        async_load_b128(dst + 64u * 4u, agg + (size_t)node * HID + f4 * 4);
    }
    #pragma unroll
    for (int i = 0; i < 16; ++i) {          // zero K=128..159 pad region
        int idx = lane + 32 * i;
        int m = idx >> 5, f = idx & 31;
        A2[m][128 + f] = 0.f;
    }
    wait_async0();
    __syncthreads();

    const v8f z8 = {0.f, 0.f, 0.f, 0.f, 0.f, 0.f, 0.f, 0.f};

    // q hidden: silu(a @ qW1 + qb1)
    {
        v8f acc[4] = {z8, z8, z8, z8};
        #pragma unroll
        for (int kc = 0; kc < 4; ++kc) {
            v16h a = a_frag_lds<160>(A2, kc * 32, halfLane, hi);
            #pragma unroll
            for (int nt = 0; nt < 4; ++nt)
                acc[nt] = wmma_fp16(a, b_frag_pk(packQ1, kc * 4 + nt, lane), acc[nt]);
        }
        #pragma unroll
        for (int nt = 0; nt < 4; ++nt) {
            int n = nt * 16 + halfLane;
            float bias = qb1[n];
            #pragma unroll
            for (int rr = 0; rr < 8; ++rr)
                HD[rr + 8 * hi][n] = silu_f(acc[nt][rr] + bias);
        }
    }
    __syncthreads();

    // q_in = hidden @ qW2 + qb2  (N=3, zero-padded packed tile)
    {
        v16h a0 = a_frag_lds<64>(HD, 0, halfLane, hi);
        v16h a1 = a_frag_lds<64>(HD, 32, halfLane, hi);
        v8f acc = z8;
        acc = wmma_fp16(a0, b_frag_pk(packQ2, 0, lane), acc);
        acc = wmma_fp16(a1, b_frag_pk(packQ2, 1, lane), acc);
        if (halfLane < 3) {
            float bias = qb2[halfLane];
            #pragma unroll
            for (int rr = 0; rr < 8; ++rr)
                QI[rr + 8 * hi][halfLane] = acc[rr] + bias;
        }
    }
    __syncthreads();

    // quantum circuit: lanes 0..15 each own one node's 8-amplitude statevector
    if (lane < 16) {
        float xin[3] = {QI[lane][0], QI[lane][1], QI[lane][2]};
        float ez[3];
        qcirc(xin, alpha_p[0], beta_p[0], gamma_p[0], delta_p[0], Lam, ez);
        A2[lane][128] = ez[0];
        A2[lane][129] = ez[1];
        A2[lane][130] = ez[2];
    }
    __syncthreads();

    // p hidden: silu(a2 @ pW1 + pb1), K padded 131 -> 160 (A pad and packed B both zero-fill)
    {
        v8f acc[4] = {z8, z8, z8, z8};
        #pragma unroll
        for (int kc = 0; kc < 5; ++kc) {
            v16h a = a_frag_lds<160>(A2, kc * 32, halfLane, hi);
            #pragma unroll
            for (int nt = 0; nt < 4; ++nt)
                acc[nt] = wmma_fp16(a, b_frag_pk(packP1, kc * 4 + nt, lane), acc[nt]);
        }
        __syncthreads();
        #pragma unroll
        for (int nt = 0; nt < 4; ++nt) {
            int n = nt * 16 + halfLane;
            float bias = pb1[n];
            #pragma unroll
            for (int rr = 0; rr < 8; ++rr)
                HD[rr + 8 * hi][n] = silu_f(acc[nt][rr] + bias);
        }
    }
    __syncthreads();

    // out = h + hidden @ pW2 + pb2
    {
        v16h a0 = a_frag_lds<64>(HD, 0, halfLane, hi);
        v16h a1 = a_frag_lds<64>(HD, 32, halfLane, hi);
        v8f acc[4] = {z8, z8, z8, z8};
        #pragma unroll
        for (int nt = 0; nt < 4; ++nt) {
            acc[nt] = wmma_fp16(a0, b_frag_pk(packP2, nt, lane), acc[nt]);
            acc[nt] = wmma_fp16(a1, b_frag_pk(packP2, 4 + nt, lane), acc[nt]);
        }
        __syncthreads();
        #pragma unroll
        for (int nt = 0; nt < 4; ++nt) {
            int n = nt * 16 + halfLane;
            float bias = pb2[n];
            #pragma unroll
            for (int rr = 0; rr < 8; ++rr)
                HD[rr + 8 * hi][n] = acc[nt][rr] + bias;
        }
        __syncthreads();
        // residual add + coalesced float4 store, masked for the grid tail
        #pragma unroll
        for (int i = 0; i < 8; ++i) {
            int idx4 = lane + 32 * i;       // 0..255 float4 units
            int m = idx4 >> 4, f4 = idx4 & 15;
            int node = nb + m;
            if (node < NN) {
                const float4 hv = *(const float4*)(h + (size_t)node * DF + f4 * 4);
                const float4 rv = *(const float4*)&HD[m][f4 * 4];
                float4 o = make_float4(hv.x + rv.x, hv.y + rv.y, hv.z + rv.z, hv.w + rv.w);
                *(float4*)(out + (size_t)node * DF + f4 * 4) = o;
            }
        }
    }
}

extern "C" void kernel_launch(void* const* d_in, const int* in_sizes, int n_in,
                              void* d_out, int out_size, void* d_ws, size_t ws_size,
                              hipStream_t stream) {
    (void)in_sizes; (void)n_in; (void)out_size; (void)ws_size;

    const float* h     = (const float*)d_in[0];
    const int*   ei    = (const int*)d_in[1];
    const float* eW1   = (const float*)d_in[2];
    const float* eb1   = (const float*)d_in[3];
    const float* eW2   = (const float*)d_in[4];
    const float* eb2   = (const float*)d_in[5];
    const float* qW1   = (const float*)d_in[6];
    const float* qb1   = (const float*)d_in[7];
    const float* qW2   = (const float*)d_in[8];
    const float* qb2   = (const float*)d_in[9];
    const float* pW1   = (const float*)d_in[10];
    const float* pb1   = (const float*)d_in[11];
    const float* pW2   = (const float*)d_in[12];
    const float* pb2   = (const float*)d_in[13];
    const float* alpha = (const float*)d_in[14];
    const float* beta  = (const float*)d_in[15];
    const float* gamma = (const float*)d_in[16];
    const float* delta = (const float*)d_in[17];
    const float* Lam   = (const float*)d_in[18];

    float* out = (float*)d_out;                    // tuple slot 0: [N,64]
    float* mij = out + (size_t)NN * DF;            // tuple slot 1: [E,64]

    char* wsb = (char*)d_ws;
    float* agg = (float*)wsb;                      // [N,64] un-normalized sums
    _Float16* packE1 = (_Float16*)(wsb + (size_t)NN * HID * 4);
    _Float16* packE2 = packE1 + 16 * FRAG_H;       // contiguous after E1 (single 24KB block)
    _Float16* packQ1 = packE2 + 8 * FRAG_H;
    _Float16* packQ2 = packQ1 + 16 * FRAG_H;
    _Float16* packP1 = packQ2 + 2 * FRAG_H;
    _Float16* packP2 = packP1 + 20 * FRAG_H;

    // weight pre-pack (tiny; runs once per call, graph-captured)
    qgcl_pack_kernel<<<16, 32, 0, stream>>>(eW1, 128, 64, 4, 1.f, 0, 0, packE1);
    qgcl_pack_kernel<<< 8, 32, 0, stream>>>(eW2,  64, 64, 4, 1.f, 0, 0, packE2);
    qgcl_pack_kernel<<<16, 32, 0, stream>>>(qW1, 128, 64, 4, NORM_INV, 64, 128, packQ1);
    qgcl_pack_kernel<<< 2, 32, 0, stream>>>(qW2,  64,  3, 1, 1.f, 0, 0, packQ2);
    qgcl_pack_kernel<<<20, 32, 0, stream>>>(pW1, 131, 64, 4, NORM_INV, 64, 128, packP1);
    qgcl_pack_kernel<<< 8, 32, 0, stream>>>(pW2,  64, 64, 4, 1.f, 0, 0, packP2);

    int n4 = NN * HID / 4;
    qgcl_zero_kernel<<<(n4 + 255) / 256, 256, 0, stream>>>((float4*)agg, n4);
    qgcl_edge_kernel<<<NE / 128, 256, 0, stream>>>(h, ei, packE1, eb1, eb2, mij, agg);
    qgcl_node_kernel<<<(NN + 63) / 64, 128, 0, stream>>>(h, agg, packQ1, qb1, packQ2, qb2,
                                                         packP1, pb1, packP2, pb2,
                                                         alpha, beta, gamma, delta, Lam, out);
}
